// Attention_70136815944325
// MI455X (gfx1250) — compile-verified, matrix operand order checked by reading, not run on hardware
//
#include <hip/hip_runtime.h>

#define NB 32
#define NT 2048
#define NF 768
#define NH 12
#define NDH 64
#define N3F 2304

typedef __attribute__((ext_vector_type(16))) __bf16 v16bf;
typedef __attribute__((ext_vector_type(8)))  float  v8f;

union ABfrag {
    v16bf v;
    unsigned int u[8];
    unsigned short s[16];
};

static __device__ __forceinline__ unsigned short f32_to_bf16(float f) {
    unsigned int u = __float_as_uint(f);
    u = (u + 0x7FFFu + ((u >> 16) & 1u)) >> 16;
    return (unsigned short)u;
}
static __device__ __forceinline__ float bf16_to_f32(unsigned short s) {
    return __uint_as_float(((unsigned int)s) << 16);
}

// element slot within a fragment-ready LDS row for K-index kk (0..31)
static __device__ __forceinline__ int frag_slot(int kk) {
    return kk + 8 * ((kk >> 3) & 1) - 8 * ((kk >> 4) & 1);
}

// ---------------------------------------------------------------- converts
__global__ __launch_bounds__(256)
void cvt_f32_bf16(const float* __restrict__ in, unsigned short* __restrict__ out, size_t n) {
    size_t i = (size_t)blockIdx.x * blockDim.x + threadIdx.x;
    size_t stride = (size_t)gridDim.x * blockDim.x;
    for (; i < n; i += stride) out[i] = f32_to_bf16(in[i]);
}

// in [R][C] f32  ->  out [C][R] bf16  (transposed)
__global__ __launch_bounds__(256)
void cvt_transpose_bf16(const float* __restrict__ in, unsigned short* __restrict__ out,
                        int R, int C) {
    size_t n = (size_t)R * C;
    size_t i = (size_t)blockIdx.x * blockDim.x + threadIdx.x;
    size_t stride = (size_t)gridDim.x * blockDim.x;
    for (; i < n; i += stride) {
        int r = (int)(i / C);
        int c = (int)(i % C);
        out[(size_t)c * R + r] = f32_to_bf16(in[i]);
    }
}

// ---------------------------------------------------------------- GEMM (async double-buffered)
// C[M,N] = A[M,K](bf16 row-major) @ Bw[N,K](bf16 pre-transposed) + bias[N]
// BM=128 BN=64 BK=32, 256 threads = 8 waves (4x2), wave tile 32x32 (2x2 wmma)
#define BM 128
#define BN 64
#define BK 32
#define LDT 40   // padded LDS row stride (elements)

template <bool OUT_BF16>
__global__ __launch_bounds__(256)
void gemm_bf16_wmma(const unsigned short* __restrict__ A,
                    const unsigned short* __restrict__ Bw,   // [N][K]
                    const float* __restrict__ bias,
                    void* __restrict__ Cv,
                    int M, int N, int K) {
    __shared__ __align__(16) unsigned short At[2][BM * LDT];
    __shared__ __align__(16) unsigned short Bt[2][BN * LDT];

    const int tid  = threadIdx.x;
    const int lane = tid & 31;
    const int wave = tid >> 5;
    const int wm   = wave >> 1;   // 0..3
    const int wn   = wave & 1;    // 0..1
    const int m_blk = blockIdx.y * BM;
    const int n_blk = blockIdx.x * BN;

    const int nin  = lane & 15;
    const int kb   = (lane >> 4) * 8;
    const int rsel = lane >> 4;

    v8f acc[2][2];
    const v8f vzero = {0.f, 0.f, 0.f, 0.f, 0.f, 0.f, 0.f, 0.f};
#pragma unroll
    for (int i = 0; i < 2; ++i)
#pragma unroll
        for (int j = 0; j < 2; ++j) acc[i][j] = vzero;

    // per-thread tile-load coordinates (fixed across K loop)
    const int ar0 = (tid) >> 2;          // A row for idx = tid
    const int ac0 = (tid & 3) * 8;
    const int ar1 = (tid + 256) >> 2;    // A row for idx = tid+256
    const int ac1 = ((tid + 256) & 3) * 8;
    const int br  = tid >> 2;            // B row (n)
    const int bc  = (tid & 3) * 8;

    unsigned ldsA0[2], ldsA1[2], ldsB[2];
#pragma unroll
    for (int s = 0; s < 2; ++s) {
        ldsA0[s] = (unsigned)(size_t)&At[s][ar0 * LDT + ac0];
        ldsA1[s] = (unsigned)(size_t)&At[s][ar1 * LDT + ac1];
        ldsB[s]  = (unsigned)(size_t)&Bt[s][br * LDT + bc];
    }

    const unsigned short* gA0 = &A[(size_t)(m_blk + ar0) * K + ac0];
    const unsigned short* gA1 = &A[(size_t)(m_blk + ar1) * K + ac1];
    const unsigned short* gB  = &Bw[(size_t)(n_blk + br) * K + bc];

    auto issue_tile = [&](int buf, int k0) {
        unsigned long long ga0 = (unsigned long long)(gA0 + k0);
        unsigned long long ga1 = (unsigned long long)(gA1 + k0);
        unsigned long long gb  = (unsigned long long)(gB + k0);
        asm volatile("global_load_async_to_lds_b128 %0, %1, off"
                     :: "v"(ldsA0[buf]), "v"(ga0) : "memory");
        asm volatile("global_load_async_to_lds_b128 %0, %1, off"
                     :: "v"(ldsA1[buf]), "v"(ga1) : "memory");
        asm volatile("global_load_async_to_lds_b128 %0, %1, off"
                     :: "v"(ldsB[buf]), "v"(gb) : "memory");
    };

    const int nk = K / BK;
    issue_tile(0, 0);

    for (int kt = 0; kt < nk; ++kt) {
        const int cur = kt & 1;
        if (kt + 1 < nk) {
            issue_tile(cur ^ 1, (kt + 1) * BK);
            // async loads retire in order: <=3 pending leaves only next tile in flight
            asm volatile("s_wait_asynccnt 0x3" ::: "memory");
        } else {
            asm volatile("s_wait_asynccnt 0x0" ::: "memory");
        }
        __syncthreads();

        ABfrag afr[2], bfr[2];
#pragma unroll
        for (int i = 0; i < 2; ++i) {
            int row = wm * 32 + i * 16 + nin;
            *reinterpret_cast<uint4*>(&afr[i].u[0]) =
                *reinterpret_cast<const uint4*>(&At[cur][row * LDT + kb]);
            *reinterpret_cast<uint4*>(&afr[i].u[4]) =
                *reinterpret_cast<const uint4*>(&At[cur][row * LDT + kb + 16]);
        }
#pragma unroll
        for (int j = 0; j < 2; ++j) {
            int col = wn * 32 + j * 16 + nin;
            *reinterpret_cast<uint4*>(&bfr[j].u[0]) =
                *reinterpret_cast<const uint4*>(&Bt[cur][col * LDT + kb]);
            *reinterpret_cast<uint4*>(&bfr[j].u[4]) =
                *reinterpret_cast<const uint4*>(&Bt[cur][col * LDT + kb + 16]);
        }
#pragma unroll
        for (int i = 0; i < 2; ++i)
#pragma unroll
            for (int j = 0; j < 2; ++j)
                acc[i][j] = __builtin_amdgcn_wmma_f32_16x16x32_bf16(
                    false, afr[i].v, false, bfr[j].v, (short)0, acc[i][j], false, false);
        __syncthreads();
    }

    // epilogue
#pragma unroll
    for (int i = 0; i < 2; ++i) {
#pragma unroll
        for (int j = 0; j < 2; ++j) {
            int mbase = m_blk + wm * 32 + i * 16 + rsel * 8;
            int col   = n_blk + wn * 32 + j * 16 + nin;
            float bv  = bias ? bias[col] : 0.f;
#pragma unroll
            for (int r = 0; r < 8; ++r) {
                float vv = acc[i][j][r] + bv;
                size_t off = (size_t)(mbase + r) * N + col;
                if (OUT_BF16)
                    ((unsigned short*)Cv)[off] = f32_to_bf16(vv);
                else
                    ((float*)Cv)[off] = vv;
            }
        }
    }
}

// ---------------------------------------------------------------- qk: S[bh][d][e] = sum_t q[t,d] k[t,e] / 8, causal mask
// grid = B*H, 128 threads = 4 waves; wave w: rows 16w..16w+15, all 64 cols
// LDS tiles stored fragment-ready: qA[d][slot], kB[e][slot] (32 slots each)
__global__ __launch_bounds__(128)
void qk_kernel(const unsigned short* __restrict__ qkv, unsigned short* __restrict__ sbuf) {
    const int bh = blockIdx.x;
    const int b = bh / NH;
    const int h = bh % NH;
    __shared__ __align__(16) unsigned short qA[64 * 32];
    __shared__ __align__(16) unsigned short kB[64 * 32];

    const int tid  = threadIdx.x;
    const int lane = tid & 31;
    const int wave = tid >> 5;
    const int nin  = lane & 15;
    const int hi16 = (lane >> 4) * 16;

    const v8f vzero = {0.f, 0.f, 0.f, 0.f, 0.f, 0.f, 0.f, 0.f};
    v8f acc[4];
#pragma unroll
    for (int j = 0; j < 4; ++j) acc[j] = vzero;

    const unsigned short* qbase = qkv + (size_t)b * NT * N3F + h * NDH;
    const unsigned short* kbase = qbase + NF;

    for (int t0 = 0; t0 < NT; t0 += 32) {
#pragma unroll
        for (int i = 0; i < 2; ++i) {
            int idx = tid + i * 128;     // 0..255
            int r = idx >> 3;            // kk within tile: 0..31
            int c = (idx & 7) * 8;       // d/e base
            int slot = frag_slot(r);
            union { uint4 q4; unsigned short s[8]; } tq, tk;
            tq.q4 = *reinterpret_cast<const uint4*>(&qbase[(size_t)(t0 + r) * N3F + c]);
            tk.q4 = *reinterpret_cast<const uint4*>(&kbase[(size_t)(t0 + r) * N3F + c]);
#pragma unroll
            for (int u = 0; u < 8; ++u) {
                qA[(c + u) * 32 + slot] = tq.s[u];
                kB[(c + u) * 32 + slot] = tk.s[u];
            }
        }
        __syncthreads();

        ABfrag afr;
        {
            int d = wave * 16 + nin;
            *reinterpret_cast<uint4*>(&afr.u[0]) =
                *reinterpret_cast<const uint4*>(&qA[d * 32 + hi16]);
            *reinterpret_cast<uint4*>(&afr.u[4]) =
                *reinterpret_cast<const uint4*>(&qA[d * 32 + hi16 + 8]);
        }
#pragma unroll
        for (int j = 0; j < 4; ++j) {
            ABfrag bfr;
            int e = j * 16 + nin;
            *reinterpret_cast<uint4*>(&bfr.u[0]) =
                *reinterpret_cast<const uint4*>(&kB[e * 32 + hi16]);
            *reinterpret_cast<uint4*>(&bfr.u[4]) =
                *reinterpret_cast<const uint4*>(&kB[e * 32 + hi16 + 8]);
            acc[j] = __builtin_amdgcn_wmma_f32_16x16x32_bf16(
                false, afr.v, false, bfr.v, (short)0, acc[j], false, false);
        }
        __syncthreads();
    }

    const int rsel = lane >> 4;
    unsigned short* sdst = sbuf + (size_t)bh * NDH * NDH;
#pragma unroll
    for (int j = 0; j < 4; ++j) {
        int e = j * 16 + nin;
#pragma unroll
        for (int r = 0; r < 8; ++r) {
            int d = wave * 16 + rsel * 8 + r;
            float v = acc[j][r] * 0.125f;       // 1/sqrt(64)
            v = (d >= e) ? v : -10000.0f;       // causal over feature axis
            sdst[d * NDH + e] = f32_to_bf16(v);
        }
    }
}

// ---------------------------------------------------------------- sv: wdt[bh][d][t] = (S @ v^T)/dk^4
// grid (T/128, B*H), 256 threads = 8 waves (2x4), wave tile 32d x 32t
__global__ __launch_bounds__(256)
void sv_kernel(const unsigned short* __restrict__ sbuf,
               const unsigned short* __restrict__ qkv,
               float* __restrict__ wdt) {
    const int bh = blockIdx.y;
    const int b = bh / NH;
    const int h = bh % NH;
    const int t_blk = blockIdx.x * 128;

    __shared__ __align__(16) unsigned short st[64 * 64];
    const int tid = threadIdx.x;
#pragma unroll
    for (int i = 0; i < 2; ++i) {
        int idx = tid + i * 256;
        *reinterpret_cast<uint4*>(&st[idx * 8]) =
            *reinterpret_cast<const uint4*>(&sbuf[(size_t)bh * 4096 + idx * 8]);
    }
    __syncthreads();

    const int lane = tid & 31;
    const int wave = tid >> 5;
    const int wm = wave >> 2;   // 0..1  (d)
    const int wt = wave & 3;    // 0..3  (t)
    const int nin = lane & 15;
    const int kb  = (lane >> 4) * 8;
    const int rsel = lane >> 4;

    const v8f vzero = {0.f, 0.f, 0.f, 0.f, 0.f, 0.f, 0.f, 0.f};
    v8f acc[2][2];
#pragma unroll
    for (int i = 0; i < 2; ++i)
#pragma unroll
        for (int j = 0; j < 2; ++j) acc[i][j] = vzero;

    const unsigned short* vbase = qkv + (size_t)b * NT * N3F + 2 * NF + h * NDH;

#pragma unroll
    for (int k0 = 0; k0 < 64; k0 += 32) {
        ABfrag afr[2];
#pragma unroll
        for (int i = 0; i < 2; ++i) {
            int d = wm * 32 + i * 16 + nin;
            *reinterpret_cast<uint4*>(&afr[i].u[0]) =
                *reinterpret_cast<const uint4*>(&st[d * 64 + k0 + kb]);
            *reinterpret_cast<uint4*>(&afr[i].u[4]) =
                *reinterpret_cast<const uint4*>(&st[d * 64 + k0 + kb + 16]);
        }
        ABfrag bfr[2];
#pragma unroll
        for (int j = 0; j < 2; ++j) {
            int t = t_blk + wt * 32 + j * 16 + nin;
            const unsigned short* vrow = vbase + (size_t)t * N3F + k0 + kb;
            *reinterpret_cast<uint4*>(&bfr[j].u[0]) = *reinterpret_cast<const uint4*>(vrow);
            *reinterpret_cast<uint4*>(&bfr[j].u[4]) = *reinterpret_cast<const uint4*>(vrow + 16);
        }
#pragma unroll
        for (int i = 0; i < 2; ++i)
#pragma unroll
            for (int j = 0; j < 2; ++j)
                acc[i][j] = __builtin_amdgcn_wmma_f32_16x16x32_bf16(
                    false, afr[i].v, false, bfr[j].v, (short)0, acc[i][j], false, false);
    }

    const float inv = 1.0f / 16777216.0f;   // 1/dk^4
#pragma unroll
    for (int i = 0; i < 2; ++i) {
#pragma unroll
        for (int j = 0; j < 2; ++j) {
            int dbase = wm * 32 + i * 16 + rsel * 8;
            int t = t_blk + wt * 32 + j * 16 + nin;
#pragma unroll
            for (int r = 0; r < 8; ++r)
                wdt[(size_t)bh * NDH * NT + (size_t)(dbase + r) * NT + t] = acc[i][j][r] * inv;
        }
    }
}

// ---------------------------------------------------------------- softmax over T + mask; writes w and a=v*w (bf16)
// one block per (b,h,d); 256 threads x 8 values, register-cached
__global__ __launch_bounds__(256)
void softmax_kernel(const float* __restrict__ wdt,
                    const unsigned short* __restrict__ qkv,
                    const float* __restrict__ mask,
                    float* __restrict__ w_out,
                    unsigned short* __restrict__ abf) {
    const int gid = blockIdx.x;
    const int d  = gid & 63;
    const int bh = gid >> 6;
    const int b = bh / NH;
    const int h = bh % NH;

    const float* row  = wdt + (size_t)bh * NDH * NT + (size_t)d * NT;
    const float* mrow = mask + (size_t)b * NT;      // mask[b,0,t,0]

    const int tid  = threadIdx.x;
    const int lane = tid & 31;
    const int wave = tid >> 5;
    __shared__ float red[8];

    float vals[8];
#pragma unroll
    for (int i = 0; i < 8; ++i) {
        int t = tid + i * 256;
        vals[i] = row[t] + mrow[t];
    }

    // max reduce
    float mx = vals[0];
#pragma unroll
    for (int i = 1; i < 8; ++i) mx = fmaxf(mx, vals[i]);
#pragma unroll
    for (int o = 16; o > 0; o >>= 1) mx = fmaxf(mx, __shfl_xor(mx, o, 32));
    if (lane == 0) red[wave] = mx;
    __syncthreads();
    mx = red[0];
#pragma unroll
    for (int w = 1; w < 8; ++w) mx = fmaxf(mx, red[w]);
    __syncthreads();

    // exp + sum reduce
    float sum = 0.f;
#pragma unroll
    for (int i = 0; i < 8; ++i) {
        vals[i] = __expf(vals[i] - mx);
        sum += vals[i];
    }
#pragma unroll
    for (int o = 16; o > 0; o >>= 1) sum += __shfl_xor(sum, o, 32);
    if (lane == 0) red[wave] = sum;
    __syncthreads();
    sum = red[0];
#pragma unroll
    for (int w = 1; w < 8; ++w) sum += red[w];
    const float inv = 1.0f / sum;

    const unsigned short* vcol = qkv + (size_t)b * NT * N3F + 2 * NF + h * NDH + d;
    const size_t obase = (size_t)b * NT * NF + h * NDH + d;
#pragma unroll
    for (int i = 0; i < 8; ++i) {
        int t = tid + i * 256;
        float p  = vals[i] * inv;
        float vv = bf16_to_f32(vcol[(size_t)t * N3F]);
        w_out[obase + (size_t)t * NF] = p;
        abf[obase + (size_t)t * NF]   = f32_to_bf16(vv * p);
    }
}

// ---------------------------------------------------------------- launch
extern "C" void kernel_launch(void* const* d_in, const int* in_sizes, int n_in,
                              void* d_out, int out_size, void* d_ws, size_t ws_size,
                              hipStream_t stream) {
    (void)in_sizes; (void)n_in; (void)out_size; (void)ws_size;
    const float* x      = (const float*)d_in[0];   // [B,T,F]
    const float* mask   = (const float*)d_in[1];   // [B,1,T,1]
    const float* W_attn = (const float*)d_in[2];   // [F,3F]
    const float* b_attn = (const float*)d_in[3];   // [1,3F]
    const float* W_proj = (const float*)d_in[4];   // [F,F]
    const float* b_proj = (const float*)d_in[5];   // [1,F]

    const size_t nX   = (size_t)NB * NT * NF;        // 50331648
    const size_t nWA  = (size_t)NF * N3F;            // 1769472
    const size_t nWP  = (size_t)NF * NF;             // 589824
    const size_t nQKV = (size_t)NB * NT * N3F;       // 150994944

    char* ws = (char*)d_ws;
    size_t off = 0;
    auto alloc = [&](size_t bytes) {
        size_t o = off;
        off += (bytes + 255) & ~(size_t)255;
        return o;
    };
    unsigned short* xbf      = (unsigned short*)(ws + alloc(nX * 2));
    unsigned short* wattn_t  = (unsigned short*)(ws + alloc(nWA * 2));   // [3F][F]
    unsigned short* wproj_t  = (unsigned short*)(ws + alloc(nWP * 2));   // [F][F] transposed
    unsigned short* qkv_bf   = (unsigned short*)(ws + alloc(nQKV * 2));
    unsigned short* sbuf     = (unsigned short*)(ws + alloc((size_t)NB * NH * NDH * NDH * 2));
    float*          wdt      = (float*)(ws + alloc(nX * 4));
    unsigned short* abf      = (unsigned short*)(ws + alloc(nX * 2));

    float* a_out = (float*)d_out;        // [B,T,F]
    float* w_out = a_out + nX;           // [B,T,F]

    cvt_f32_bf16<<<4096, 256, 0, stream>>>(x, xbf, nX);
    cvt_transpose_bf16<<<1024, 256, 0, stream>>>(W_attn, wattn_t, NF, N3F);
    cvt_transpose_bf16<<<1024, 256, 0, stream>>>(W_proj, wproj_t, NF, NF);

    // qkv = x @ W_attn + b_attn   -> bf16
    gemm_bf16_wmma<true><<<dim3(N3F / BN, (NB * NT) / BM), 256, 0, stream>>>(
        xbf, wattn_t, b_attn, (void*)qkv_bf, NB * NT, N3F, NF);

    // S = (q^T k)/sqrt(dk), masked -> bf16
    qk_kernel<<<NB * NH, 128, 0, stream>>>(qkv_bf, sbuf);

    // wdt = (S @ v^T)/dk^4
    sv_kernel<<<dim3(NT / 128, NB * NH), 256, 0, stream>>>(sbuf, qkv_bf, wdt);

    // softmax over T + mask; writes w_out and abf = v*w
    softmax_kernel<<<NB * NH * NDH, 256, 0, stream>>>(wdt, qkv_bf, mask, w_out, abf);

    // a = abf @ W_proj + b_proj   -> f32
    gemm_bf16_wmma<false><<<dim3(NF / BN, (NB * NT) / BM), 256, 0, stream>>>(
        abf, wproj_t, b_proj, (void*)a_out, NB * NT, NF, NF);
}
